// TreeLSTMLevelEncoder_25323127177874
// MI455X (gfx1250) — matrix operand order repaired and checked by reading, not compiled, not run on hardware
//
#include <hip/hip_runtime.h>
#include <hip/hip_bf16.h>
#include <math.h>

// ---------------------------------------------------------------------------
// Child-sum TreeLSTM level encoder for gfx1250 (MI455X).
// f16 WMMA (v_wmma_f32_16x16x32_f16) with f32 accumulation.
// - Weights pre-packed once per call to f16 column-major (~3.7MB, L2-resident);
//   B fragments load straight from global (2x global_load_b128 per frag).
// - A tiles staged in LDS via vector ds_store_b128 (f32->f16 conversion).
// - Deterministic child gathers (index-ordered scan within each tree block).
// ---------------------------------------------------------------------------

typedef _Float16 half8_t __attribute__((ext_vector_type(8)));
typedef _Float16 v16h    __attribute__((ext_vector_type(16)));
typedef float    v8f     __attribute__((ext_vector_type(8)));
typedef float    f4      __attribute__((ext_vector_type(4)));

#define MAXL 8   // generator: T=50, K=4 => max depth 3; 8 gives margin

__device__ __forceinline__ v8f wmma_f16(v16h a, v16h b, v8f c) {
  return __builtin_amdgcn_wmma_f32_16x16x32_f16(
      /*neg_a=*/false, a, /*neg_b=*/false, b,
      /*c_mod=*/(short)0, c, /*reuse_a=*/false, /*reuse_b=*/false);
}

// A-matrix 16x32 f16 fragment (ISA 7.12.2): lane r (=lane&15) holds row r;
// hi=lane>>4 selects K ranges {hi*8..hi*8+7} (v0-3) and {16+hi*8..} (v4-7).
__device__ __forceinline__ v16h ld_afrag(const _Float16* row, int hi) {
  half8_t lo = *(const half8_t*)(row + hi * 8);
  half8_t hh = *(const half8_t*)(row + 16 + hi * 8);
  return __builtin_shufflevector(lo, hh, 0,1,2,3,4,5,6,7,8,9,10,11,12,13,14,15);
}
// B-matrix 32x16 f16 fragment: lane col = lane&15; hi=lane>>4 selects K
// ranges {hi*8..hi*8+7} and {16+hi*8..} (same as A). p points at (col, kbase)
// in a column-major f16 weight image.
__device__ __forceinline__ v16h ld_bfrag(const _Float16* p, int hi) {
  half8_t lo = *(const half8_t*)(p + hi * 8);
  half8_t hh = *(const half8_t*)(p + 16 + hi * 8);
  return __builtin_shufflevector(lo, hh, 0,1,2,3,4,5,6,7,8,9,10,11,12,13,14,15);
}

__device__ __forceinline__ float sigmoidf_(float x) { return 1.0f / (1.0f + __expf(-x)); }

// convert 16 consecutive f32 -> two half8 vector LDS stores (ds_store_b128)
__device__ __forceinline__ void cvt_store16(_Float16* dst, const float* src) {
  f4 v0 = *(const f4*)(src + 0);
  f4 v1 = *(const f4*)(src + 4);
  f4 v2 = *(const f4*)(src + 8);
  f4 v3 = *(const f4*)(src + 12);
  half8_t a, b;
  a[0]=(_Float16)v0.x; a[1]=(_Float16)v0.y; a[2]=(_Float16)v0.z; a[3]=(_Float16)v0.w;
  a[4]=(_Float16)v1.x; a[5]=(_Float16)v1.y; a[6]=(_Float16)v1.z; a[7]=(_Float16)v1.w;
  b[0]=(_Float16)v2.x; b[1]=(_Float16)v2.y; b[2]=(_Float16)v2.z; b[3]=(_Float16)v2.w;
  b[4]=(_Float16)v3.x; b[5]=(_Float16)v3.y; b[6]=(_Float16)v3.z; b[7]=(_Float16)v3.w;
  *(half8_t*)(dst)     = a;
  *(half8_t*)(dst + 8) = b;
}

// ---------------------------------------------------------------------------
__global__ void k_init_h(float* __restrict__ h, long long NH) {
  long long i = (long long)blockIdx.x * blockDim.x + threadIdx.x;
  long long stride = (long long)gridDim.x * blockDim.x;
  for (long long j = i; j < NH; j += stride) h[j] = 0.f;
}

__global__ void k_zero_counts(int* __restrict__ counts) {
  if (threadIdx.x < 32) counts[threadIdx.x] = 0;
}

// build per-level node lists (set deterministic; order irrelevant: rows are
// written to node-indexed locations and sums are gathered in index order)
__global__ void k_build(const int* __restrict__ depth, int* __restrict__ counts,
                        int* __restrict__ lists, int N) {
  int i = blockIdx.x * blockDim.x + threadIdx.x;
  if (i < N) {
    int d = depth[i];
    if (d < MAXL) {
      int pos = atomicAdd(&counts[d], 1);
      lists[(size_t)d * N + pos] = i;
    }
  }
}

// pack [[W_f];[U_f]] -> Wtf[col][k] f16 and [[W_iou];[U_iou]] -> Wtiou[col][k]
__global__ void k_packW(const float* __restrict__ Wf, const float* __restrict__ Uf,
                        const float* __restrict__ Wiou, const float* __restrict__ Uiou,
                        _Float16* __restrict__ Wtf, _Float16* __restrict__ Wtiou,
                        int Din, int H, int Ktot) {
  long long totalF = (long long)H * Ktot;
  long long total  = totalF * 4;               // H*Ktot + 3H*Ktot
  long long i = (long long)blockIdx.x * blockDim.x + threadIdx.x;
  long long stride = (long long)gridDim.x * blockDim.x;
  int H3 = 3 * H;
  for (long long idx = i; idx < total; idx += stride) {
    if (idx < totalF) {
      int col = (int)(idx / Ktot), k = (int)(idx % Ktot);
      float v = (k < Din) ? Wf[(size_t)k * H + col] : Uf[(size_t)(k - Din) * H + col];
      Wtf[idx] = (_Float16)v;
    } else {
      long long j = idx - totalF;
      int col = (int)(j / Ktot), k = (int)(j % Ktot);
      float v = (k < Din) ? Wiou[(size_t)k * H3 + col] : Uiou[(size_t)(k - Din) * H3 + col];
      Wtiou[j] = (_Float16)v;
    }
  }
}

// ---------------------------------------------------------------------------
// Kernel A: per-child forget gate
//   fc_msg[n] = sigmoid([embed[parent[n]] | h[n]] . [[W_f];[U_f]] + b_f) * c[n]
// rows = nodes at depth lvl+1. Tile 128x64, K=Din+H, 8 waves (4M x 2N),
// per-wave 2x2 WMMA tiles; B frags from packed global weights.
// ---------------------------------------------------------------------------
__global__ __launch_bounds__(256)
void k_childgate(const float* __restrict__ embed, const _Float16* __restrict__ Wtf,
                 const float* __restrict__ bf,
                 const int* __restrict__ parent, const int* __restrict__ list,
                 const int* __restrict__ cnt,
                 const float* __restrict__ h, const float* __restrict__ cst,
                 float* __restrict__ fc_msg, int Din, int H) {
  const int TM = 128, TN = 64, KS = 32;
  __shared__ __align__(16) _Float16 As[2][TM][KS];
  __shared__ int nodeS[TM];
  __shared__ int erowS[TM];

  int count = *cnt;
  int nColT = H / TN;
  int mTile = blockIdx.x / nColT;
  int nTile = blockIdx.x % nColT;
  if (mTile * TM >= count) return;

  int tid = threadIdx.x;
  if (tid < TM) {
    int gi = mTile * TM + tid;
    int node = (gi < count) ? list[gi] : -1;
    nodeS[tid] = node;
    erowS[tid] = (node >= 0) ? parent[node] : -1;
  }
  __syncthreads();

  const int Ktot = Din + H;
  const int NK = Ktot / KS;

  auto load_stage = [&](int ks, int buf) {
    int kbase = ks * KS;
    int rl = tid >> 1;
    int c0 = (tid & 1) * 16;
    _Float16* dst = &As[buf][rl][c0];
    int node = nodeS[rl];
    if (node >= 0) {
      const float* src = (kbase < Din)
          ? embed + (size_t)erowS[rl] * Din + kbase + c0
          : h + (size_t)node * H + (kbase - Din) + c0;
      cvt_store16(dst, src);
    } else {
      half8_t z = {};
      *(half8_t*)(dst) = z;
      *(half8_t*)(dst + 8) = z;
    }
  };

  int wid = tid >> 5, lane = tid & 31;
  int hi = lane >> 4, lc = lane & 15;
  int waveM = wid & 3, waveN = wid >> 2;
  int mb = waveM * 32, nb = waveN * 32;
  const _Float16* wcol0 = Wtf + (size_t)(nTile * TN + nb + lc) * Ktot;
  const _Float16* wcol1 = Wtf + (size_t)(nTile * TN + nb + 16 + lc) * Ktot;

  v8f acc[2][2] = {};
  load_stage(0, 0);
  __syncthreads();
  for (int ks = 0; ks < NK; ++ks) {
    int buf = ks & 1;
    if (ks + 1 < NK) load_stage(ks + 1, buf ^ 1);
    int kb = ks * KS;
    v16h bfr0 = ld_bfrag(wcol0 + kb, hi);
    v16h bfr1 = ld_bfrag(wcol1 + kb, hi);
    v16h afr0 = ld_afrag(&As[buf][mb + lc][0], hi);
    v16h afr1 = ld_afrag(&As[buf][mb + 16 + lc][0], hi);
    acc[0][0] = wmma_f16(afr0, bfr0, acc[0][0]);
    acc[0][1] = wmma_f16(afr0, bfr1, acc[0][1]);
    acc[1][0] = wmma_f16(afr1, bfr0, acc[1][0]);
    acc[1][1] = wmma_f16(afr1, bfr1, acc[1][1]);
    __syncthreads();
  }

  // epilogue: f = sigmoid(acc + b_f); fc_msg = f * c
#pragma unroll
  for (int mi = 0; mi < 2; ++mi) {
#pragma unroll
    for (int ni = 0; ni < 2; ++ni) {
      int col = nTile * TN + nb + ni * 16 + lc;
      float bias = bf[col];
#pragma unroll
      for (int i = 0; i < 8; ++i) {
        int rl = mb + mi * 16 + hi * 8 + i;
        int gi = mTile * TM + rl;
        if (gi < count) {
          int node = nodeS[rl];
          size_t off = (size_t)node * H + col;
          float fv = sigmoidf_(acc[mi][ni][i] + bias);
          fc_msg[off] = fv * cst[off];
        }
      }
    }
  }
}

// ---------------------------------------------------------------------------
// Kernel B1: deterministic child gather (persistent). One loop slot per parent
// at depth lvl: scan the T nodes of its tree in index order.
// ---------------------------------------------------------------------------
__global__ __launch_bounds__(256)
void k_gather(const float* __restrict__ h, const float* __restrict__ fc_msg,
              const int* __restrict__ parent, const int* __restrict__ depth,
              const int* __restrict__ list, const int* __restrict__ cnt,
              float* __restrict__ hsum, float* __restrict__ fcsum,
              int H, int T, int childLvl) {
  int cntv = *cnt;
  __shared__ int flags[256];
  int t = threadIdx.x;
  for (int b = blockIdx.x; b < cntv; b += gridDim.x) {
    int p = list[b];
    int base = (p / T) * T;
    __syncthreads();
    if (t < T) flags[t] = (parent[base + t] == p && depth[base + t] == childLvl) ? 1 : 0;
    __syncthreads();
    for (int col = t; col < H; col += 256) {
      float hs = 0.f, fs = 0.f;
      for (int tt = 0; tt < T; ++tt) {       // fixed index order => deterministic
        if (flags[tt]) {
          size_t off = (size_t)(base + tt) * H + col;
          hs += h[off];
          fs += fc_msg[off];
        }
      }
      size_t po = (size_t)p * H + col;
      hsum[po] = hs;
      fcsum[po] = fs;
    }
  }
}

// ---------------------------------------------------------------------------
// Kernel B2: node update (3 gates fused so the epilogue sees i,o,u together)
//   [i|o|u] = [embed[p] | hsum[p]] . [[W_iou];[U_iou]] + b_iou
//   c = sig(i)*tanh(u) + fcsum[p];  h = sig(o)*tanh(c)
// Tile 128 rows x 32 cols x 3 gates; 8 waves (4M x 2N), 2x1x3 WMMA per wave.
// ---------------------------------------------------------------------------
__global__ __launch_bounds__(256)
void k_update(const float* __restrict__ embed, const _Float16* __restrict__ Wtiou,
              const float* __restrict__ biou,
              const int* __restrict__ list, const int* __restrict__ cnt,
              const float* __restrict__ hsum, const float* __restrict__ fcsum,
              float* __restrict__ h, float* __restrict__ c, int Din, int H) {
  const int TM = 128, TN = 32, KS = 32;
  __shared__ __align__(16) _Float16 As[2][TM][KS];
  __shared__ int nodeS[TM];

  int count = *cnt;
  int nColT = H / TN;
  int mTile = blockIdx.x / nColT;
  int nTile = blockIdx.x % nColT;
  if (mTile * TM >= count) return;

  int tid = threadIdx.x;
  if (tid < TM) {
    int gi = mTile * TM + tid;
    nodeS[tid] = (gi < count) ? list[gi] : -1;
  }
  __syncthreads();

  const int Ktot = Din + H;
  const int NK = Ktot / KS;

  auto load_stage = [&](int ks, int buf) {
    int kbase = ks * KS;
    int rl = tid >> 1;
    int c0 = (tid & 1) * 16;
    _Float16* dst = &As[buf][rl][c0];
    int node = nodeS[rl];
    if (node >= 0) {
      const float* src = (kbase < Din)
          ? embed + (size_t)node * Din + kbase + c0
          : hsum + (size_t)node * H + (kbase - Din) + c0;
      cvt_store16(dst, src);
    } else {
      half8_t z = {};
      *(half8_t*)(dst) = z;
      *(half8_t*)(dst + 8) = z;
    }
  };

  int wid = tid >> 5, lane = tid & 31;
  int hi = lane >> 4, lc = lane & 15;
  int waveM = wid & 3, waveN = wid >> 2;
  int mb = waveM * 32, nb = waveN * 16;
  int col = nTile * TN + nb + lc;             // 0..H-1
  const _Float16* wc0 = Wtiou + (size_t)(0 * H + col) * Ktot;
  const _Float16* wc1 = Wtiou + (size_t)(1 * H + col) * Ktot;
  const _Float16* wc2 = Wtiou + (size_t)(2 * H + col) * Ktot;

  v8f acc[2][3] = {};
  load_stage(0, 0);
  __syncthreads();
  for (int ks = 0; ks < NK; ++ks) {
    int buf = ks & 1;
    if (ks + 1 < NK) load_stage(ks + 1, buf ^ 1);
    int kb = ks * KS;
    v16h b0 = ld_bfrag(wc0 + kb, hi);
    v16h b1 = ld_bfrag(wc1 + kb, hi);
    v16h b2 = ld_bfrag(wc2 + kb, hi);
    v16h a0 = ld_afrag(&As[buf][mb + lc][0], hi);
    v16h a1 = ld_afrag(&As[buf][mb + 16 + lc][0], hi);
    acc[0][0] = wmma_f16(a0, b0, acc[0][0]);
    acc[0][1] = wmma_f16(a0, b1, acc[0][1]);
    acc[0][2] = wmma_f16(a0, b2, acc[0][2]);
    acc[1][0] = wmma_f16(a1, b0, acc[1][0]);
    acc[1][1] = wmma_f16(a1, b1, acc[1][1]);
    acc[1][2] = wmma_f16(a1, b2, acc[1][2]);
    __syncthreads();
  }

  float bi = biou[col];
  float bo = biou[col + H];
  float bu = biou[col + 2 * H];
#pragma unroll
  for (int mi = 0; mi < 2; ++mi) {
#pragma unroll
    for (int i = 0; i < 8; ++i) {
      int rl = mb + mi * 16 + hi * 8 + i;
      int gi = mTile * TM + rl;
      if (gi < count) {
        int node = nodeS[rl];
        size_t off = (size_t)node * H + col;
        float iv = sigmoidf_(acc[mi][0][i] + bi);
        float ov = sigmoidf_(acc[mi][1][i] + bo);
        float uv = tanhf(acc[mi][2][i] + bu);
        float cn = iv * uv + fcsum[off];
        float hn = ov * tanhf(cn);
        c[off] = cn;
        h[off] = hn;
      }
    }
  }
}

// ---------------------------------------------------------------------------
// Readout: per-graph sum of h (index-ordered => deterministic), split + tanh.
// out = [mu (B x H/2) | tanh(logvar) (B x H/2)] flat.
// ---------------------------------------------------------------------------
__global__ __launch_bounds__(256)
void k_readout(const float* __restrict__ h, float* __restrict__ out,
               int H, int T, int Bnum) {
  int b = blockIdx.x;
  int H2 = H / 2;
  size_t base = (size_t)b * T;
  for (int col = threadIdx.x; col < H; col += 256) {
    float s = 0.f;
    for (int r = 0; r < T; ++r) s += h[(base + r) * H + col];
    if (col < H2) out[(size_t)b * H2 + col] = s;
    else out[(size_t)Bnum * H2 + (size_t)b * H2 + (col - H2)] = tanhf(s);
  }
}

// ---------------------------------------------------------------------------
extern "C" void kernel_launch(void* const* d_in, const int* in_sizes, int n_in,
                              void* d_out, int out_size, void* d_ws, size_t ws_size,
                              hipStream_t stream) {
  const float* embed = (const float*)d_in[0];
  const float* W_iou = (const float*)d_in[1];
  const float* U_iou = (const float*)d_in[2];
  const float* b_iou = (const float*)d_in[3];
  const float* W_f   = (const float*)d_in[4];
  const float* U_f   = (const float*)d_in[5];
  const float* b_f   = (const float*)d_in[6];
  const int* parent  = (const int*)d_in[7];
  const int* depth   = (const int*)d_in[8];

  const int N    = in_sizes[7];          // 100000 nodes
  const int H    = in_sizes[6];          // 512 (b_f length)
  const int Din  = in_sizes[0] / N;      // 384
  const int Bnum = out_size / H;         // 2000 graphs
  const int T    = N / Bnum;             // 50 nodes per tree (contiguous)
  const int Ktot = Din + H;              // 896

  // workspace: h, c, fc_msg, hsum, fcsum (f32), counts, lists, packed weights
  float* h     = (float*)d_ws;
  float* c     = h    + (size_t)N * H;
  float* fc    = c    + (size_t)N * H;
  float* hsum  = fc   + (size_t)N * H;
  float* fcsum = hsum + (size_t)N * H;
  int* counts  = (int*)(fcsum + (size_t)N * H);
  int* lists   = counts + 32;
  _Float16* Wtf   = (_Float16*)(lists + (size_t)MAXL * N);
  _Float16* Wtiou = Wtf + (size_t)H * Ktot;

  k_init_h<<<4096, 256, 0, stream>>>(h, (long long)N * H);
  k_zero_counts<<<1, 32, 0, stream>>>(counts);
  k_packW<<<4096, 256, 0, stream>>>(W_f, U_f, W_iou, U_iou, Wtf, Wtiou, Din, H, Ktot);
  k_build<<<(N + 255) / 256, 256, 0, stream>>>(depth, counts, lists, N);

  const int mT = (N + 127) / 128;
  dim3 gA((unsigned)(mT * (H / 64)));
  dim3 gB2((unsigned)(mT * (H / 32)));

  for (int lvl = MAXL - 1; lvl >= 0; --lvl) {
    const int* clist = lists + (size_t)(lvl + 1) * N;   // children (depth lvl+1)
    const int* ccnt  = counts + (lvl + 1);              // counts[MAXL]==0 guard
    const int* plist = lists + (size_t)lvl * N;         // nodes at this level
    const int* pcnt  = counts + lvl;

    k_childgate<<<gA, 256, 0, stream>>>(embed, Wtf, b_f, parent,
                                        clist, ccnt, h, c, fc, Din, H);
    k_gather<<<8192, 256, 0, stream>>>(h, fc, parent, depth, plist, pcnt,
                                       hsum, fcsum, H, T, lvl + 1);
    k_update<<<gB2, 256, 0, stream>>>(embed, Wtiou, b_iou, plist, pcnt,
                                      hsum, fcsum, h, c, Din, H);
  }

  k_readout<<<Bnum, 256, 0, stream>>>(h, (float*)d_out, H, T, Bnum);
}
